// TemporalSelfAttention_33122787787413
// MI455X (gfx1250) — matrix operand order repaired and checked by reading, not compiled, not run on hardware
//
#include <hip/hip_runtime.h>
#include <math.h>

typedef __attribute__((ext_vector_type(16))) _Float16 v16h;
typedef __attribute__((ext_vector_type(8)))  float    v8f;

#define N_ 32
#define C_ 64
#define T_ 256
#define V_ 25
#define NPAIR (N_ * V_)            // 800 (n,v) pairs

// LDS layout (dynamic shared memory, byte offsets). Pitches in f16 elements.
#define XS 72
#define WS 72
#define QS 72
#define X_OFF   0                          // X tile:   256 x 72 f16 = 36864 B
#define W_OFF   (T_ * XS * 2)              // W tiles: 3 x 64 x 72 f16 = 27648 B
#define Q_OFF   (W_OFF + 3 * C_ * WS * 2)  // Q/K/V:   3 x 256 x 72 f16 = 110592 B
#define LDS_TOTAL (Q_OFF + 3 * T_ * QS * 2)   // 175104 B (< 320 KB WGP LDS)
// Phase-2 scratch aliases the (dead) X region:
#define SL_OFF  0                          // S repack: 8 waves x 16 x 40 f16 = 10240 B
#define PS_OFF  10240                      // per-lane sum partials: 8*32*4 f32 = 4096 B
#define PQ_OFF  (10240 + 4096)             // per-lane sumsq partials

__global__ __launch_bounds__(256)
void tsa_attn_kernel(const float* __restrict__ x,
                     const float* __restrict__ Wq, const float* __restrict__ bq,
                     const float* __restrict__ Wk, const float* __restrict__ bk,
                     const float* __restrict__ Wv, const float* __restrict__ bv,
                     float* __restrict__ out,
                     float* __restrict__ pairSum, float* __restrict__ pairSq)
{
    extern __shared__ char smem[];
    _Float16* Xl = (_Float16*)(smem + X_OFF);
    _Float16* Wl = (_Float16*)(smem + W_OFF);
    _Float16* Ql = (_Float16*)(smem + Q_OFF);
    _Float16* Kl = Ql + T_ * QS;
    _Float16* Vl = Kl + T_ * QS;
    _Float16* Sl = (_Float16*)(smem + SL_OFF);
    float*    pS = (float*)(smem + PS_OFF);
    float*    pQ = (float*)(smem + PQ_OFF);

    const int tid  = threadIdx.x;
    const int wv   = tid >> 5;      // wave id 0..7 (wave32)
    const int lane = tid & 31;
    const int hi   = lane >> 4;     // lane half (WMMA operand layout)
    const int n16  = lane & 15;
    const int pair = blockIdx.x;    // (n,v) pair
    const int nIdx = pair / V_;
    const int vIdx = pair - nIdx * V_;

    // ---- Phase 0: stage X(t,c) and the three W(d,c) into LDS as f16 ----
    for (int i = tid; i < T_ * C_; i += 256) {
        int t = i & (T_ - 1);
        int c = i >> 8;
        Xl[t * XS + c] = (_Float16)x[((nIdx * C_ + c) * T_ + t) * V_ + vIdx];
    }
    for (int i = tid; i < 3 * C_ * C_; i += 256) {
        int m = i >> 12;
        int r = i & 4095;
        int d = r >> 6;
        int c = r & 63;
        const float* Wm = (m == 0) ? Wq : ((m == 1) ? Wk : Wv);
        Wl[(m * C_ + d) * WS + c] = (_Float16)Wm[d * C_ + c];
    }
    __syncthreads();

    // ---- Phase 1: Q/K/V = X * W^T + b  (192 16x16 tiles, K=64 -> 2 WMMAs) ----
    for (int tile = wv; tile < 192; tile += 8) {
        int m  = tile / 64;              // 0:Q 1:K 2:V
        int rm = tile - m * 64;
        int rt = rm >> 2;                // row tile 0..15 (t)
        int dt = rm & 3;                 // col tile 0..3  (d)
        v8f acc = {0.f,0.f,0.f,0.f,0.f,0.f,0.f,0.f};
        #pragma unroll
        for (int kt = 0; kt < 2; ++kt) {
            v16h a, b;
            #pragma unroll
            for (int i = 0; i < 16; ++i) {
                int Ka = hi * 16 + i; // A-frag K map
                a[i] = Xl[(rt * 16 + n16) * XS + kt * 32 + Ka];
                // B[k=c][n=d] = W[d][c]; b[i] = B[hi*16+i][n16]
                b[i] = Wl[(m * C_ + dt * 16 + n16) * WS + kt * 32 + hi * 16 + i];
            }
            acc = __builtin_amdgcn_wmma_f32_16x16x32_f16(false, a, false, b,
                                                         (short)0, acc, false, false);
        }
        const float* bm = (m == 0) ? bq : ((m == 1) ? bk : bv);
        float bb = bm[dt * 16 + n16];
        _Float16* Dst = (m == 0) ? Ql : ((m == 1) ? Kl : Vl);
        #pragma unroll
        for (int j = 0; j < 8; ++j)   // D layout: VGPR j -> M = j + hi*8, N = n16
            Dst[(rt * 16 + j + hi * 8) * QS + dt * 16 + n16] = (_Float16)(acc[j] + bb);
    }
    __syncthreads();

    // ---- Phase 2: S = tanh(Q K^T / 8); O = S * V; per-channel stats ----
    float rSum[4] = {0.f, 0.f, 0.f, 0.f};
    float rSq[4]  = {0.f, 0.f, 0.f, 0.f};
    for (int rb = 0; rb < 2; ++rb) {
        const int sb = wv + rb * 8;     // this wave's 16-row s-block (tile index)
        v8f o[4];
        #pragma unroll
        for (int dt = 0; dt < 4; ++dt) {
            v8f z = {0.f,0.f,0.f,0.f,0.f,0.f,0.f,0.f};
            o[dt] = z;
        }
        for (int tt = 0; tt < 8; ++tt) {            // t in blocks of 32
            #pragma unroll
            for (int u = 0; u < 2; ++u) {           // two 16x16 S tiles
                const int tT = tt * 2 + u;
                v8f sa = {0.f,0.f,0.f,0.f,0.f,0.f,0.f,0.f};
                #pragma unroll
                for (int kt = 0; kt < 2; ++kt) {    // K = C = 64
                    v16h a, b;
                    #pragma unroll
                    for (int i = 0; i < 16; ++i) {
                        int Ka = hi * 16 + i;
                        a[i] = Ql[(sb * 16 + n16) * QS + kt * 32 + Ka];
                        // B[k=c][n=t'] = K[t'][c]
                        b[i] = Kl[(tT * 16 + n16) * QS + kt * 32 + hi * 16 + i];
                    }
                    sa = __builtin_amdgcn_wmma_f32_16x16x32_f16(false, a, false, b,
                                                                (short)0, sa, false, false);
                }
                #pragma unroll
                for (int j = 0; j < 8; ++j)  // tanh + 1/sqrt(C); repack via LDS
                    Sl[(wv * 16 + j + hi * 8) * 40 + u * 16 + n16] =
                        (_Float16)tanhf(sa[j] * 0.125f);
            }
            __syncthreads();
            v16h a;
            #pragma unroll
            for (int i = 0; i < 16; ++i) {
                int Ka = hi * 16 + i;
                a[i] = Sl[(wv * 16 + n16) * 40 + Ka];   // S as A-fragment (K = t)
            }
            #pragma unroll
            for (int dt = 0; dt < 4; ++dt) {
                v16h b;
                #pragma unroll
                for (int i = 0; i < 16; ++i)            // B[k=t][n=d] = V[t][d]
                    b[i] = Vl[(tt * 32 + hi * 16 + i) * QS + dt * 16 + n16];
                o[dt] = __builtin_amdgcn_wmma_f32_16x16x32_f16(false, a, false, b,
                                                               (short)0, o[dt], false, false);
            }
            __syncthreads();
        }
        // write pre-norm output (n,c,t,v) and accumulate per-channel stats
        #pragma unroll
        for (int dt = 0; dt < 4; ++dt) {
            const int ch = dt * 16 + n16;
            #pragma unroll
            for (int j = 0; j < 8; ++j) {
                const int sRow = sb * 16 + j + hi * 8;
                float val = o[dt][j];
                out[((nIdx * C_ + ch) * T_ + sRow) * V_ + vIdx] = val;
                rSum[dt] += val;
                rSq[dt]  += val * val;
            }
        }
    }
    // deterministic per-pair channel reduction (no float atomics)
    #pragma unroll
    for (int dt = 0; dt < 4; ++dt) {
        pS[(wv * 32 + lane) * 4 + dt] = rSum[dt];
        pQ[(wv * 32 + lane) * 4 + dt] = rSq[dt];
    }
    __syncthreads();
    if (tid < C_) {
        const int dt = tid >> 4, nn = tid & 15;
        float s1 = 0.f, s2 = 0.f;
        #pragma unroll
        for (int w = 0; w < 8; ++w) {
            s1 += pS[(w * 32 + nn) * 4 + dt] + pS[(w * 32 + 16 + nn) * 4 + dt];
            s2 += pQ[(w * 32 + nn) * 4 + dt] + pQ[(w * 32 + 16 + nn) * 4 + dt];
        }
        pairSum[pair * C_ + tid] = s1;
        pairSq [pair * C_ + tid] = s2;
    }
}

__global__ void tsa_stats_kernel(const float* __restrict__ pairSum,
                                 const float* __restrict__ pairSq,
                                 const float* __restrict__ gamma,
                                 float* __restrict__ meanArr,
                                 float* __restrict__ scaleArr)
{
    int c = threadIdx.x;  // 64 threads, one per channel
    float s1 = 0.f, s2 = 0.f;
    for (int p = 0; p < NPAIR; ++p) {
        s1 += pairSum[p * C_ + c];
        s2 += pairSq [p * C_ + c];
    }
    const float inv = 1.0f / (float)(N_ * T_ * V_);
    float mean = s1 * inv;
    float var  = s2 * inv - mean * mean;
    meanArr[c]  = mean;
    scaleArr[c] = rsqrtf(var + 1e-5f) * gamma[c];
}

__global__ __launch_bounds__(256)
void tsa_norm_kernel(float* __restrict__ out,
                     const float* __restrict__ meanArr,
                     const float* __restrict__ scaleArr,
                     const float* __restrict__ beta)
{
    int idx = blockIdx.x * 256 + threadIdx.x;     // total divisible by 256
    int c = (idx / (T_ * V_)) & (C_ - 1);
    out[idx] = (out[idx] - meanArr[c]) * scaleArr[c] + beta[c];
}

extern "C" void kernel_launch(void* const* d_in, const int* in_sizes, int n_in,
                              void* d_out, int out_size, void* d_ws, size_t ws_size,
                              hipStream_t stream)
{
    (void)in_sizes; (void)n_in; (void)out_size; (void)ws_size;
    const float* x     = (const float*)d_in[0];
    const float* Wq    = (const float*)d_in[1];
    const float* bq    = (const float*)d_in[2];
    const float* Wk    = (const float*)d_in[3];
    const float* bk    = (const float*)d_in[4];
    const float* Wv    = (const float*)d_in[5];
    const float* bv    = (const float*)d_in[6];
    const float* gamma = (const float*)d_in[7];
    const float* beta  = (const float*)d_in[8];
    float* out = (float*)d_out;

    float* pairSum  = (float*)d_ws;                 // 800*64 f32
    float* pairSq   = pairSum + NPAIR * C_;         // 800*64 f32
    float* meanArr  = pairSq  + NPAIR * C_;         // 64 f32
    float* scaleArr = meanArr + C_;                 // 64 f32

    (void)hipFuncSetAttribute((const void*)tsa_attn_kernel,
                              hipFuncAttributeMaxDynamicSharedMemorySize, LDS_TOTAL);

    tsa_attn_kernel<<<NPAIR, 256, LDS_TOTAL, stream>>>(x, Wq, bq, Wk, bk, Wv, bv,
                                                       out, pairSum, pairSq);
    tsa_stats_kernel<<<1, C_, 0, stream>>>(pairSum, pairSq, gamma, meanArr, scaleArr);
    tsa_norm_kernel<<<(N_ * C_ * T_ * V_) / 256, 256, 0, stream>>>(out, meanArr,
                                                                   scaleArr, beta);
}
